// SelfAttention_58119497450330
// MI455X (gfx1250) — matrix operand order repaired and checked by reading, not compiled, not run on hardware
//
#include <hip/hip_runtime.h>
#include <hip/hip_bf16.h>

typedef __attribute__((ext_vector_type(16))) _Float16 v16h;
typedef __attribute__((ext_vector_type(8)))  _Float16 v8h;
typedef __attribute__((ext_vector_type(8)))  float    v8f;

#define EMBED 768
#define NHEAD 12
#define HEADD 64
#define SEQ   1024
#define BATCH 8
#define POS   256
#define MTOT  (BATCH * SEQ)   // 8192

// ---------------------------------------------------------------------------
// fp32 -> fp16 conversion
// ---------------------------------------------------------------------------
__global__ __launch_bounds__(256) void cvt_f32_f16(const float* __restrict__ in,
                                                   _Float16* __restrict__ out, int n) {
    int i = blockIdx.x * blockDim.x + threadIdx.x;
    if (i < n) out[i] = (_Float16)in[i];
}

// ---------------------------------------------------------------------------
// WMMA fragment helpers (layouts per CDNA5 ISA 7.12.2, wave32)
// ---------------------------------------------------------------------------
// A (16x32 f16): lane l holds row (l&15); halves j=0..7 -> K=k0+8*(l>>4)+j,
// j=8..15 -> K=k0+16+8*(l>>4)+(j-8). Two contiguous 16B chunks per lane.
__device__ __forceinline__ v16h load_afrag(const _Float16* __restrict__ base,
                                           int ld, int m0, int k0, int lane) {
    const int lo = lane & 15, hi = lane >> 4;
    const _Float16* p = base + (size_t)(m0 + lo) * ld + k0 + 8 * hi;
    v8h c0 = *(const v8h*)p;
    v8h c1 = *(const v8h*)(p + 16);
    v16h a;
#pragma unroll
    for (int i = 0; i < 8; ++i) { a[i] = c0[i]; a[8 + i] = c1[i]; }
    return a;
}

// B (32x16 f16) from [N,K] row-major storage (i.e. W with y = x @ W^T):
// lane l holds column n0+(l&15); j=0..7 -> K=k0+8*(l>>4)+j,
// j=8..15 -> K=k0+16+8*(l>>4)+(j-8).
__device__ __forceinline__ v16h load_bfrag(const _Float16* __restrict__ base,
                                           int ld, int n0, int k0, int lane) {
    const int lo = lane & 15, hi = lane >> 4;
    const _Float16* p = base + (size_t)(n0 + lo) * ld + k0 + 8 * hi;
    v8h c0 = *(const v8h*)p;
    v8h c1 = *(const v8h*)(p + 16);
    v16h b;
#pragma unroll
    for (int i = 0; i < 8; ++i) { b[i] = c0[i]; b[8 + i] = c1[i]; }
    return b;
}

__device__ __forceinline__ v8f wmma_f16(v16h a, v16h b, v8f c) {
    return __builtin_amdgcn_wmma_f32_16x16x32_f16(false, a, false, b, (short)0, c,
                                                  false, false);
}

// ---------------------------------------------------------------------------
// Fused QKV projection:  {Q,K,V} = x @ {Wq,Wk,Wv}^T.  The A matrix (x) is
// shared by all three GEMMs, so each A fragment is loaded once per k-chunk
// and consumed by 12 WMMAs (4 per weight) -> 3x less A traffic.
//  Q,K stored [B,H,S,D] f16; V stored transposed [B,H,D,S] f16 so the PV
//  WMMA B-fragment in the attention kernel is two contiguous 16-byte loads.
// Block = 128 threads (4 waves); wave w: rows [m0,m0+16) x 64 cols.
// ---------------------------------------------------------------------------
__global__ __launch_bounds__(128)
void qkv_gemm(const _Float16* __restrict__ A,
              const _Float16* __restrict__ Wq, const _Float16* __restrict__ Wk,
              const _Float16* __restrict__ Wv,
              _Float16* __restrict__ qh, _Float16* __restrict__ kh,
              _Float16* __restrict__ vth) {
    const int lane = threadIdx.x & 31;
    const int w    = threadIdx.x >> 5;
    const int lo = lane & 15, hi = lane >> 4;
    const int m0 = blockIdx.y * 64 + w * 16;
    const int n0 = blockIdx.x * 64;

    v8f accq[4], acck[4], accv[4];
#pragma unroll
    for (int nn = 0; nn < 4; ++nn) {
        accq[nn] = (v8f){}; acck[nn] = (v8f){}; accv[nn] = (v8f){};
    }

    for (int k = 0; k < EMBED; k += 32) {
        if (k + 32 < EMBED)
            __builtin_prefetch(A + (size_t)(m0 + lo) * EMBED + k + 32, 0, 3);
        const v16h a = load_afrag(A, EMBED, m0, k, lane);
        v16h bfr[4];
#pragma unroll
        for (int nn = 0; nn < 4; ++nn) bfr[nn] = load_bfrag(Wq, EMBED, n0 + nn * 16, k, lane);
#pragma unroll
        for (int nn = 0; nn < 4; ++nn) accq[nn] = wmma_f16(a, bfr[nn], accq[nn]);
#pragma unroll
        for (int nn = 0; nn < 4; ++nn) bfr[nn] = load_bfrag(Wk, EMBED, n0 + nn * 16, k, lane);
#pragma unroll
        for (int nn = 0; nn < 4; ++nn) acck[nn] = wmma_f16(a, bfr[nn], acck[nn]);
#pragma unroll
        for (int nn = 0; nn < 4; ++nn) bfr[nn] = load_bfrag(Wv, EMBED, n0 + nn * 16, k, lane);
#pragma unroll
        for (int nn = 0; nn < 4; ++nn) accv[nn] = wmma_f16(a, bfr[nn], accv[nn]);
    }

#pragma unroll
    for (int nn = 0; nn < 4; ++nn) {
        const int n = n0 + nn * 16 + lo, h = n >> 6, d = n & 63;
#pragma unroll
        for (int r = 0; r < 8; ++r) {
            const int m = m0 + r + 8 * hi, b = m >> 10, s = m & 1023;
            qh [(size_t)((b * NHEAD + h) * SEQ + s) * HEADD + d] = (_Float16)accq[nn][r];
            kh [(size_t)((b * NHEAD + h) * SEQ + s) * HEADD + d] = (_Float16)acck[nn][r];
            vth[(size_t)((b * NHEAD + h) * HEADD + d) * SEQ + s] = (_Float16)accv[nn][r];
        }
    }
}

// ---------------------------------------------------------------------------
// Output projection: out = y[M,K] @ Wp[N,K]^T, fp32 store row-major.
// ---------------------------------------------------------------------------
__global__ __launch_bounds__(128)
void out_gemm(const _Float16* __restrict__ A, const _Float16* __restrict__ W,
              float* __restrict__ outf) {
    const int lane = threadIdx.x & 31;
    const int w    = threadIdx.x >> 5;
    const int lo = lane & 15, hi = lane >> 4;
    const int m0 = blockIdx.y * 64 + w * 16;
    const int n0 = blockIdx.x * 64;

    v8f acc[4];
#pragma unroll
    for (int nn = 0; nn < 4; ++nn) acc[nn] = (v8f){};

    for (int k = 0; k < EMBED; k += 32) {
        if (k + 32 < EMBED)
            __builtin_prefetch(A + (size_t)(m0 + lo) * EMBED + k + 32, 0, 3);
        const v16h a = load_afrag(A, EMBED, m0, k, lane);
        v16h bfr[4];
#pragma unroll
        for (int nn = 0; nn < 4; ++nn) bfr[nn] = load_bfrag(W, EMBED, n0 + nn * 16, k, lane);
#pragma unroll
        for (int nn = 0; nn < 4; ++nn) acc[nn] = wmma_f16(a, bfr[nn], acc[nn]);
    }

#pragma unroll
    for (int nn = 0; nn < 4; ++nn)
#pragma unroll
        for (int r = 0; r < 8; ++r)
            outf[(size_t)(m0 + r + 8 * hi) * EMBED + n0 + nn * 16 + lo] = acc[nn][r];
}

// ---------------------------------------------------------------------------
// Fused flash attention with relative-position bias.
// Grid: (SEQ/64, BATCH*NHEAD), block 128 (4 waves); each wave = 16 queries.
// ---------------------------------------------------------------------------
__global__ __launch_bounds__(128)
void attn_wmma(const _Float16* __restrict__ qh,   // [B,H,S,D]
               const _Float16* __restrict__ kh,   // [B,H,S,D]
               const _Float16* __restrict__ vt,   // [B,H,D,S]
               const _Float16* __restrict__ posh, // [POS,D]
               _Float16* __restrict__ yh) {       // [B,S,E]
    __shared__ _Float16 qpos[4][16][POS];   // 32 KB: per-wave Q @ pos^T bias table
    __shared__ _Float16 pstage[4][16][64];  //  8 KB: C-layout -> A-layout transpose

    const int lane = threadIdx.x & 31;
    const int w    = threadIdx.x >> 5;
    const int lo = lane & 15, hi = lane >> 4;

    const int bh = blockIdx.y;
    const int b  = bh / NHEAD;
    const int h  = bh % NHEAD;
    const int q0 = blockIdx.x * 64 + w * 16;

    const _Float16* qbh = qh + (size_t)bh * SEQ * HEADD;
    const _Float16* kbh = kh + (size_t)bh * SEQ * HEADD;
    const _Float16* vbh = vt + (size_t)bh * HEADD * SEQ;

    // resident Q fragments (D=64 -> two K=32 chunks)
    v16h aq[2];
#pragma unroll
    for (int kc = 0; kc < 2; ++kc) aq[kc] = load_afrag(qbh, HEADD, q0, kc * 32, lane);

    // bias table: qpos[row][p] = q[row] . pos[p]
    for (int pt = 0; pt < POS / 16; ++pt) {
        v8f acc = (v8f){};
#pragma unroll
        for (int kc = 0; kc < 2; ++kc) {
            v16h bp = load_bfrag(posh, HEADD, pt * 16, kc * 32, lane);
            acc = wmma_f16(aq[kc], bp, acc);
        }
#pragma unroll
        for (int r = 0; r < 8; ++r)
            qpos[w][r + 8 * hi][pt * 16 + lo] = (_Float16)acc[r];
    }

    float mrow[8], lrow[8];
    v8f o[4];
#pragma unroll
    for (int r = 0; r < 8; ++r) { mrow[r] = -1e30f; lrow[r] = 0.f; }
#pragma unroll
    for (int d = 0; d < 4; ++d) o[d] = (v8f){};

    for (int jt = 0; jt < SEQ / 64; ++jt) {
        // S = Q K^T  (4 col subtiles of 16 keys)
        v8f sc[4];
        v16h bk[4][2];
#pragma unroll
        for (int jj = 0; jj < 4; ++jj)
#pragma unroll
            for (int kc = 0; kc < 2; ++kc)
                bk[jj][kc] = load_bfrag(kbh, HEADD, jt * 64 + jj * 16, kc * 32, lane);
#pragma unroll
        for (int jj = 0; jj < 4; ++jj) {
            sc[jj] = (v8f){};
#pragma unroll
            for (int kc = 0; kc < 2; ++kc) sc[jj] = wmma_f16(aq[kc], bk[jj][kc], sc[jj]);
        }
        // add relative-position bias, scale, tile row-max
        float tmax[8];
#pragma unroll
        for (int r = 0; r < 8; ++r) tmax[r] = -1e30f;
#pragma unroll
        for (int jj = 0; jj < 4; ++jj) {
            const int j = jt * 64 + jj * 16 + lo;
#pragma unroll
            for (int r = 0; r < 8; ++r) {
                const int i = q0 + r + 8 * hi;
                int rel = j - i;
                rel = rel < -128 ? -128 : (rel > 127 ? 127 : rel);
                const float bias = (float)qpos[w][r + 8 * hi][rel & 255];
                const float vsc = (sc[jj][r] + bias) * 0.125f;
                sc[jj][r] = vsc;
                tmax[r] = fmaxf(tmax[r], vsc);
            }
        }
        // row reduction across the 16 lanes holding one row
#pragma unroll
        for (int r = 0; r < 8; ++r)
#pragma unroll
            for (int mk = 1; mk < 16; mk <<= 1)
                tmax[r] = fmaxf(tmax[r], __shfl_xor(tmax[r], mk, 32));
        // online softmax update
        float alpha[8], rs[8];
#pragma unroll
        for (int r = 0; r < 8; ++r) {
            const float mnew = fmaxf(mrow[r], tmax[r]);
            alpha[r] = __expf(mrow[r] - mnew);
            mrow[r] = mnew;
            rs[r] = 0.f;
        }
#pragma unroll
        for (int jj = 0; jj < 4; ++jj)
#pragma unroll
            for (int r = 0; r < 8; ++r) {
                const float p = __expf(sc[jj][r] - mrow[r]);
                sc[jj][r] = p;
                rs[r] += p;
            }
#pragma unroll
        for (int r = 0; r < 8; ++r) {
#pragma unroll
            for (int mk = 1; mk < 16; mk <<= 1) rs[r] += __shfl_xor(rs[r], mk, 32);
            lrow[r] = lrow[r] * alpha[r] + rs[r];
        }
#pragma unroll
        for (int d = 0; d < 4; ++d)
#pragma unroll
            for (int r = 0; r < 8; ++r) o[d][r] *= alpha[r];

        // transpose P through LDS: C-layout (lane=col) -> A-layout (lane=row).
        // Same-wave LDS ops are in-order on CDNA5, no barrier needed.
#pragma unroll
        for (int jj = 0; jj < 4; ++jj)
#pragma unroll
            for (int r = 0; r < 8; ++r)
                pstage[w][r + 8 * hi][jj * 16 + lo] = (_Float16)sc[jj][r];
        v16h ap[2];
#pragma unroll
        for (int kc = 0; kc < 2; ++kc) {
            const _Float16* pp = &pstage[w][lo][kc * 32 + 8 * hi];
#pragma unroll
            for (int i = 0; i < 8; ++i) { ap[kc][i] = pp[i]; ap[kc][8 + i] = pp[16 + i]; }
        }
        // O += P @ V  (V stored transposed [D,S] -> contiguous B fragments)
        v16h bv[4][2];
#pragma unroll
        for (int d = 0; d < 4; ++d)
#pragma unroll
            for (int kc = 0; kc < 2; ++kc)
                bv[d][kc] = load_bfrag(vbh, SEQ, d * 16, jt * 64 + kc * 32, lane);
#pragma unroll
        for (int d = 0; d < 4; ++d)
#pragma unroll
            for (int kc = 0; kc < 2; ++kc)
                o[d] = wmma_f16(ap[kc], bv[d][kc], o[d]);
    }

    // normalize, merge heads: yh[b, s, h*64 + d]
#pragma unroll
    for (int d = 0; d < 4; ++d)
#pragma unroll
        for (int r = 0; r < 8; ++r) {
            const float v = o[d][r] / lrow[r];
            yh[(size_t)(b * SEQ + q0 + r + 8 * hi) * EMBED + h * HEADD + d * 16 + lo] =
                (_Float16)v;
        }
}

// ---------------------------------------------------------------------------
extern "C" void kernel_launch(void* const* d_in, const int* in_sizes, int n_in,
                              void* d_out, int out_size, void* d_ws, size_t ws_size,
                              hipStream_t stream) {
    (void)in_sizes; (void)n_in; (void)out_size; (void)ws_size;
    const float* x   = (const float*)d_in[0];
    const float* Wq  = (const float*)d_in[1];
    const float* Wk  = (const float*)d_in[2];
    const float* Wv  = (const float*)d_in[3];
    const float* Wp  = (const float*)d_in[4];
    const float* pos = (const float*)d_in[5];

    const size_t NX = (size_t)MTOT * EMBED;     // 6291456
    const size_t NW = (size_t)EMBED * EMBED;    // 589824
    const size_t NP = (size_t)POS * HEADD;      // 16384

    _Float16* p = (_Float16*)d_ws;
    _Float16* xh   = p; p += NX;
    _Float16* wqh  = p; p += NW;
    _Float16* wkh  = p; p += NW;
    _Float16* wvh  = p; p += NW;
    _Float16* wph  = p; p += NW;
    _Float16* posh = p; p += NP;
    _Float16* qh   = p; p += NX;   // [B,H,S,D]
    _Float16* kh   = p; p += NX;   // [B,H,S,D]
    _Float16* vth  = p; p += NX;   // [B,H,D,S]
    _Float16* yh   = p; p += NX;   // [B,S,E]

    // 1. precision conversion
    cvt_f32_f16<<<dim3((NX + 255) / 256), 256, 0, stream>>>(x,   xh,   (int)NX);
    cvt_f32_f16<<<dim3((NW + 255) / 256), 256, 0, stream>>>(Wq,  wqh,  (int)NW);
    cvt_f32_f16<<<dim3((NW + 255) / 256), 256, 0, stream>>>(Wk,  wkh,  (int)NW);
    cvt_f32_f16<<<dim3((NW + 255) / 256), 256, 0, stream>>>(Wv,  wvh,  (int)NW);
    cvt_f32_f16<<<dim3((NW + 255) / 256), 256, 0, stream>>>(Wp,  wph,  (int)NW);
    cvt_f32_f16<<<dim3((NP + 255) / 256), 256, 0, stream>>>(pos, posh, (int)NP);

    // 2. fused QKV projection (A fragments shared across the 3 GEMMs)
    dim3 ggrid(EMBED / 64, MTOT / 64);          // (12, 128)
    qkv_gemm<<<ggrid, 128, 0, stream>>>(xh, wqh, wkh, wvh, qh, kh, vth);

    // 3. fused attention (WMMA flash loop + rel-pos bias)
    attn_wmma<<<dim3(SEQ / 64, BATCH * NHEAD), 128, 0, stream>>>(qh, kh, vth, posh, yh);

    // 4. output projection -> fp32 d_out
    out_gemm<<<ggrid, 128, 0, stream>>>(yh, wph, (float*)d_out);
}
